// Custom_Model_Qlinear_27968827031786
// MI455X (gfx1250) — 8-bit matrix operand order repaired and checked by reading, compiled, not run on hardware
//
#include <hip/hip_runtime.h>
#include <hip/hip_bf16.h>


typedef __attribute__((ext_vector_type(8))) int v8i;
typedef __attribute__((ext_vector_type(4))) int i4v;
typedef __attribute__((ext_vector_type(2))) int i2v;

#define M_DIM 8192
#define K_DIM 4096
#define N_DIM 4096
#define KD     (K_DIM / 4)
#define KSTEPS (K_DIM / 64)
#define LDSTR  20

#if __has_builtin(__builtin_amdgcn_global_load_async_to_lds_b128)
#define HAVE_ASYNC 1
#else
#define HAVE_ASYNC 0
#endif

typedef __attribute__((address_space(1))) i4v* as1_p4;
typedef __attribute__((address_space(3))) i4v* as3_p4;

__device__ __forceinline__ void wait_async0() {
#if __has_builtin(__builtin_amdgcn_s_wait_asynccnt)
  __builtin_amdgcn_s_wait_asynccnt(0);
#else
  asm volatile("s_wait_asynccnt 0x0" ::: "memory");
#endif
}

#if HAVE_ASYNC
__device__ __forceinline__ void async_ld16(const int* g, int* l) {
  __builtin_amdgcn_global_load_async_to_lds_b128((as1_p4)(i4v*)(int*)g,
                                                 (as3_p4)(i4v*)l, 0, 0);
}
#endif

__global__ __launch_bounds__(256) void pack_i8(const int* __restrict__ src,
                                               int* __restrict__ dst,
                                               int n_dwords) {
  int t = blockIdx.x * blockDim.x + threadIdx.x;
  int base = t * 4;
  if (base >= n_dwords) return;
  const i4v* s = (const i4v*)src;
  i4v o;
#pragma unroll
  for (int j = 0; j < 4; ++j) {
    i4v v = s[base + j];
    o[j] = (v[0] & 0xFF) | ((v[1] & 0xFF) << 8) |
           ((v[2] & 0xFF) << 16) | (v[3] << 24);
  }
  *(i4v*)(dst + base) = o;
}

__global__ __launch_bounds__(256) void qgemm_i8_wmma(
    const int* __restrict__ A, const int* __restrict__ B,
    const float* __restrict__ bias, const float* __restrict__ iscale,
    const float* __restrict__ wscale, float* __restrict__ out) {
  __shared__ int ldsA[2][128 * LDSTR];
  __shared__ int ldsB[2][128 * LDSTR];

  const int tid  = threadIdx.x;
  const int lane = tid & 31;
  const int wave = tid >> 5;
  const int wm   = (wave & 1) * 64;
  const int wn   = (wave >> 1) * 32;

  const int bm = blockIdx.y * 128;
  const int bn = blockIdx.x * 128;

  const int ldRow   = tid >> 2;
  const int ldChunk = (tid & 3) * 4;
  const int ldsOff0 = ldRow * LDSTR + ldChunk;
  const int ldsOff1 = (ldRow + 64) * LDSTR + ldChunk;

  const int* pa0 = A + (bm + ldRow) * KD + ldChunk;
  const int* pa1 = pa0 + 64 * KD;
  const int* pb0 = B + (bn + ldRow) * KD + ldChunk;
  const int* pb1 = pb0 + 64 * KD;

  const int l15 = lane & 15;
  const int lhi = lane >> 4;

  v8i acc[4][2] = {};

#if HAVE_ASYNC
  async_ld16(pa0, &ldsA[0][ldsOff0]);
  async_ld16(pa1, &ldsA[0][ldsOff1]);
  async_ld16(pb0, &ldsB[0][ldsOff0]);
  async_ld16(pb1, &ldsB[0][ldsOff1]);
#else
  i4v ra0 = *(const i4v*)pa0;
  i4v ra1 = *(const i4v*)pa1;
  i4v rb0 = *(const i4v*)pb0;
  i4v rb1 = *(const i4v*)pb1;
#endif
  pa0 += 16; pa1 += 16; pb0 += 16; pb1 += 16;

  for (int kk = 0; kk < KSTEPS; ++kk) {
    const int cur = kk & 1;
#if HAVE_ASYNC
    wait_async0();
    __syncthreads();
    if (kk + 1 < KSTEPS) {
      const int nxt = cur ^ 1;
      async_ld16(pa0, &ldsA[nxt][ldsOff0]);
      async_ld16(pa1, &ldsA[nxt][ldsOff1]);
      async_ld16(pb0, &ldsB[nxt][ldsOff0]);
      async_ld16(pb1, &ldsB[nxt][ldsOff1]);
      pa0 += 16; pa1 += 16; pb0 += 16; pb1 += 16;
    }
#else
    *(i4v*)(&ldsA[cur][ldsOff0]) = ra0;
    *(i4v*)(&ldsA[cur][ldsOff1]) = ra1;
    *(i4v*)(&ldsB[cur][ldsOff0]) = rb0;
    *(i4v*)(&ldsB[cur][ldsOff1]) = rb1;
    __syncthreads();
    if (kk + 1 < KSTEPS) {
      ra0 = *(const i4v*)pa0;
      ra1 = *(const i4v*)pa1;
      rb0 = *(const i4v*)pb0;
      rb1 = *(const i4v*)pb1;
      pa0 += 16; pa1 += 16; pb0 += 16; pb1 += 16;
    }
#endif

    v8i bf[2];
#pragma unroll
    for (int nt = 0; nt < 2; ++nt) {
      const int* p = &ldsB[cur][(wn + nt * 16 + l15) * LDSTR + lhi * 4];
      i4v q0 = *(const i4v*)(p);
      i4v q1 = *(const i4v*)(p + 8);
      bf[nt] = (v8i){q0[0], q0[1], q0[2], q0[3], q1[0], q1[1], q1[2], q1[3]};
    }
#pragma unroll
    for (int mt = 0; mt < 4; ++mt) {
      const int* p = &ldsA[cur][(wm + mt * 16 + l15) * LDSTR + lhi * 4];
      i2v q0 = *(const i2v*)(p);
      i2v q1 = *(const i2v*)(p + 2);
      i2v q2 = *(const i2v*)(p + 8);
      i2v q3 = *(const i2v*)(p + 10);
      v8i af = (v8i){q0[0], q0[1], q1[0], q1[1], q2[0], q2[1], q3[0], q3[1]};
      acc[mt][0] = __builtin_amdgcn_wmma_i32_16x16x64_iu8(
          true, af, true, bf[0], acc[mt][0], false, false);
      acc[mt][1] = __builtin_amdgcn_wmma_i32_16x16x64_iu8(
          true, af, true, bf[1], acc[mt][1], false, false);
    }
  }

  const float si = iscale[0];
#pragma unroll
  for (int nt = 0; nt < 2; ++nt) {
    const int col  = bn + wn + nt * 16 + l15;
    const float cs = si * wscale[col];
    const float bv = bias[col];
#pragma unroll
    for (int mt = 0; mt < 4; ++mt) {
      const int row0 = bm + wm + mt * 16 + lhi * 8;
#pragma unroll
      for (int r = 0; r < 8; ++r) {
        out[(size_t)(row0 + r) * N_DIM + col] =
            (float)acc[mt][nt][r] * cs + bv;
      }
    }
  }
}

extern "C" void kernel_launch(void* const* d_in, const int* in_sizes, int n_in,
                              void* d_out, int out_size, void* d_ws,
                              size_t ws_size, hipStream_t stream) {
  const int*   inp  = (const int*)d_in[0];
  const int*   wgt  = (const int*)d_in[1];
  const float* bias = (const float*)d_in[2];
  const float* isc  = (const float*)d_in[3];
  const float* wsc  = (const float*)d_in[5];
  float* out = (float*)d_out;

  int* Apk = (int*)d_ws;
  int* Bpk = (int*)((char*)d_ws + (size_t)M_DIM * K_DIM);

  {
    int nd = M_DIM * K_DIM / 4;
    pack_i8<<<(nd / 4) / 256, 256, 0, stream>>>(inp, Apk, nd);
  }
  {
    int nd = N_DIM * K_DIM / 4;
    pack_i8<<<(nd / 4) / 256, 256, 0, stream>>>(wgt, Bpk, nd);
  }

  dim3 grid(N_DIM / 128, M_DIM / 128);
  qgemm_i8_wmma<<<grid, 256, 0, stream>>>(Apk, Bpk, bias, isc, wsc, out);
}
